// EncoderLayer_45363444580696
// MI455X (gfx1250) — compile-verified, matrix operand order checked by reading, not run on hardware
//
#include <hip/hip_runtime.h>
#include <hip/hip_bf16.h>
#include <math.h>

#define BB   2
#define TT   2048
#define DM   1024
#define NH   16
#define HD   64
#define DFF  4096

typedef __bf16 bf16_t;
typedef __attribute__((ext_vector_type(16))) __bf16 bf16x16;
typedef __attribute__((ext_vector_type(8)))  float  floatx8;
typedef unsigned int u32x4 __attribute__((ext_vector_type(4)));
typedef int          i32x8 __attribute__((ext_vector_type(8)));
typedef int          i32x4 __attribute__((ext_vector_type(4)));

__device__ __forceinline__ floatx8 wmma_bf16(bf16x16 a, bf16x16 b, floatx8 c) {
  // D = A(16x32 bf16) * B(32x16 bf16) + C(16x16 f32)
  return __builtin_amdgcn_wmma_f32_16x16x32_bf16(false, a, false, b, (short)0, c, false, false);
}

// ---- TDM: async-load a 2D tile (tileRows x 32 bf16 elements) into LDS -------
// Builds the Tensor DMA Descriptor (CDNA5 ISA) in SGPRs.
__device__ __forceinline__ void tdm_load_2d(const bf16_t* gtile, unsigned ldsOff,
                                            unsigned rowLen /*tensor_dim0, elems*/,
                                            unsigned nRows  /*tensor_dim1*/,
                                            unsigned tileRows) {
  unsigned long long ga = (unsigned long long)(uintptr_t)gtile;
  // Group0: count=1 | lds_addr | global_addr[56:0] | type=2 (bits 127:126)
  u32x4 g0 = { 1u,
               ldsOff,
               (unsigned)(ga & 0xffffffffull),
               (unsigned)((ga >> 32) & 0x01ffffffull) | 0x80000000u };
  // Group1: data_size=1 (2 bytes); tensor_dim0/1; tile_dim0=32; tile_dim1=tileRows;
  //         tensor_dim0_stride=rowLen; tensor_dim1_stride=0
  i32x8 g1 = { (int)(1u << 16),
               (int)((rowLen & 0xffffu) << 16),
               (int)((rowLen >> 16) | ((nRows & 0xffffu) << 16)),
               (int)(((nRows >> 16) & 0xffffu) | (32u << 16)),
               (int)(tileRows & 0xffffu),
               (int)rowLen,
               0, 0 };
  i32x4 z4 = { 0, 0, 0, 0 };
  i32x8 z8 = { 0, 0, 0, 0, 0, 0, 0, 0 };
  __builtin_amdgcn_tensor_load_to_lds(g0, g1, z4, z4, z8, 0);
}

// ---------------- fp32 -> bf16 transpose of weights: Wt[n*K+k] = W[k*N+n] ----
__global__ void transpose_bf16_kernel(const float* __restrict__ W, bf16_t* __restrict__ Wt,
                                      int K, int N) {
  long idx = (long)blockIdx.x * blockDim.x + threadIdx.x;
  long total = (long)K * N;
  if (idx >= total) return;
  int n = (int)(idx / K), k = (int)(idx % K);
  Wt[idx] = (bf16_t)W[(long)k * N + n];
}

// ---------------- LayerNorm (wave per row of DM=1024) -> bf16 ----------------
__global__ void ln_bf16_kernel(const float* __restrict__ x, const float* __restrict__ w,
                               const float* __restrict__ b, bf16_t* __restrict__ out, int rows) {
  int wave = (int)((blockIdx.x * (long)blockDim.x + threadIdx.x) >> 5);
  int lane = threadIdx.x & 31;
  if (wave >= rows) return;
  const float* px = x + (long)wave * DM;
  float vals[32];
  float s = 0.f, ss = 0.f;
#pragma unroll
  for (int i = 0; i < 32; ++i) { float v = px[lane + i * 32]; vals[i] = v; s += v; ss += v * v; }
#pragma unroll
  for (int o = 16; o; o >>= 1) { s += __shfl_xor(s, o, 32); ss += __shfl_xor(ss, o, 32); }
  float mu = s * (1.f / DM);
  float rs = rsqrtf(ss * (1.f / DM) - mu * mu + 1e-5f);
  bf16_t* po = out + (long)wave * DM;
#pragma unroll
  for (int i = 0; i < 32; ++i) {
    int c = lane + i * 32;
    po[c] = (bf16_t)((vals[i] - mu) * rs * w[c] + b[c]);
  }
}

// ------- bf16 WMMA GEMM, TDM double-buffered LDS staging, wave tile 16x64 ----
// Block: 256 threads (8 waves), block tile 32 (M) x 256 (N), K step 32.
// EPI: 0=bf16 store, 1=GELU->bf16, 2=residual+bias->f32, 3=QKV scatter [B,H,T,hd],
//      4=V scatter transposed [B,H,hd,T]
template <int EPI>
__global__ void gemm_bf16_kernel(const bf16_t* __restrict__ A, const bf16_t* __restrict__ Bt,
                                 const float* __restrict__ bias, const float* __restrict__ res,
                                 float* __restrict__ outF, bf16_t* __restrict__ outB,
                                 int M, int N, int K) {
  __shared__ __align__(64) bf16_t ldsA[2][32 * 32];    //  2 KB x2
  __shared__ __align__(64) bf16_t ldsB[2][256 * 32];   // 16 KB x2

  int wid = threadIdx.x >> 5, lane = threadIdx.x & 31;
  int mblk = blockIdx.y * 32;
  int nblk = blockIdx.x * 256;
  int m0 = mblk + (wid >> 2) * 16;
  int n0loc = (wid & 3) * 64;

  bool issuer = (threadIdx.x < 32);   // wave 0 drives the TDM (EXEC-independent issue)
  if (issuer) {
    tdm_load_2d(A  + (long)mblk * K, (unsigned)(uintptr_t)&ldsA[0][0], (unsigned)K, (unsigned)M, 32u);
    tdm_load_2d(Bt + (long)nblk * K, (unsigned)(uintptr_t)&ldsB[0][0], (unsigned)K, (unsigned)N, 256u);
  }

  floatx8 acc[4] = {};
  int arow = (wid >> 2) * 16 + (lane & 15);
  int kh = (lane >> 4) * 16;
  int buf = 0;
  for (int k0 = 0; k0 < K; k0 += 32) {
    bool has_next = (k0 + 32 < K);
    if (issuer) {
      if (has_next) {
        tdm_load_2d(A  + (long)mblk * K + k0 + 32, (unsigned)(uintptr_t)&ldsA[buf ^ 1][0],
                    (unsigned)K, (unsigned)M, 32u);
        tdm_load_2d(Bt + (long)nblk * K + k0 + 32, (unsigned)(uintptr_t)&ldsB[buf ^ 1][0],
                    (unsigned)K, (unsigned)N, 256u);
        __builtin_amdgcn_s_wait_tensorcnt(2);  // current buffer's 2 DMAs retired
      } else {
        __builtin_amdgcn_s_wait_tensorcnt(0);
      }
    }
    __syncthreads();                            // LDS tiles visible to all 8 waves
    bf16x16 a = *(const bf16x16*)(&ldsA[buf][arow * 32 + kh]);
#pragma unroll
    for (int j = 0; j < 4; ++j) {
      bf16x16 bt = *(const bf16x16*)(&ldsB[buf][(n0loc + j * 16 + (lane & 15)) * 32 + kh]);
      acc[j] = wmma_bf16(a, bt, acc[j]);
    }
    __syncthreads();                            // done reading before next overwrite
    buf ^= 1;
  }

#pragma unroll
  for (int j = 0; j < 4; ++j) {
    int n = nblk + n0loc + j * 16 + (lane & 15);
    float bb = bias[n];
#pragma unroll
    for (int e = 0; e < 8; ++e) {
      int m = m0 + e + (lane >> 4) * 8;
      float v = acc[j][e] + bb;
      if (EPI == 0) {
        outB[(long)m * N + n] = (bf16_t)v;
      } else if (EPI == 1) {
        outB[(long)m * N + n] = (bf16_t)(0.5f * v * (1.f + erff(v * 0.70710678f)));
      } else if (EPI == 2) {
        outF[(long)m * N + n] = res[(long)m * N + n] + v;
      } else if (EPI == 3) {
        int bq = m / TT, t = m % TT, hh = n / HD, d = n % HD;
        outB[(((long)(bq * NH + hh) * TT) + t) * HD + d] = (bf16_t)v;
      } else if (EPI == 4) {
        int bq = m / TT, t = m % TT, hh = n / HD, d = n % HD;
        outB[(((long)(bq * NH + hh) * HD) + d) * TT + t] = (bf16_t)v;
      }
    }
  }
}

// ---------------- scores = (Q K^T)/8, masked, -> d_out attn region -----------
__global__ void scores_kernel(const bf16_t* __restrict__ q, const bf16_t* __restrict__ k,
                              const unsigned char* __restrict__ mask, float* __restrict__ attn) {
  int bh = blockIdx.z;
  int wid = threadIdx.x >> 5, lane = threadIdx.x & 31;
  int m0 = blockIdx.y * 32 + (wid >> 2) * 16;
  int n0 = blockIdx.x * 256 + (wid & 3) * 64;
  const bf16_t* qb = q + (long)bh * TT * HD;
  const bf16_t* kb = k + (long)bh * TT * HD;
  floatx8 acc[4] = {};
  int ar = m0 + (lane & 15);
  int kh = (lane >> 4) * 16;
#pragma unroll
  for (int k0 = 0; k0 < HD; k0 += 32) {
    bf16x16 a = *(const bf16x16*)(qb + (long)ar * HD + k0 + kh);
#pragma unroll
    for (int j = 0; j < 4; ++j) {
      bf16x16 bt = *(const bf16x16*)(kb + (long)(n0 + j * 16 + (lane & 15)) * HD + k0 + kh);
      acc[j] = wmma_bf16(a, bt, acc[j]);
    }
  }
  const unsigned char* mb = mask + (long)bh * TT * TT;
  float* ab = attn + (long)bh * TT * TT;
#pragma unroll
  for (int j = 0; j < 4; ++j) {
    int n = n0 + j * 16 + (lane & 15);
#pragma unroll
    for (int e = 0; e < 8; ++e) {
      int m = m0 + e + (lane >> 4) * 8;
      float v = acc[j][e] * 0.125f;
      if (mb[(long)m * TT + n]) v = -1e10f;
      ab[(long)m * TT + n] = v;
    }
  }
}

// ---------------- in-place row softmax over T=2048 (wave per row) ------------
__global__ void softmax_kernel(float* __restrict__ attn, int rows) {
  int wave = (int)((blockIdx.x * (long)blockDim.x + threadIdx.x) >> 5);
  int lane = threadIdx.x & 31;
  if (wave >= rows) return;
  float* p = attn + (long)wave * TT;
  float v[64];
  float mx = -3.4e38f;
#pragma unroll
  for (int i = 0; i < 64; ++i) { v[i] = p[lane + i * 32]; mx = fmaxf(mx, v[i]); }
#pragma unroll
  for (int o = 16; o; o >>= 1) mx = fmaxf(mx, __shfl_xor(mx, o, 32));
  float s = 0.f;
#pragma unroll
  for (int i = 0; i < 64; ++i) { v[i] = __expf(v[i] - mx); s += v[i]; }
#pragma unroll
  for (int o = 16; o; o >>= 1) s += __shfl_xor(s, o, 32);
  float inv = 1.f / s;
#pragma unroll
  for (int i = 0; i < 64; ++i) p[lane + i * 32] = v[i] * inv;
}

// ---------------- ctx = attn(f32, cvt->bf16) @ V (bf16, pre-transposed) ------
__global__ void ctx_kernel(const float* __restrict__ attn, const bf16_t* __restrict__ vT,
                           bf16_t* __restrict__ ctx) {
  int bh = blockIdx.z;
  int wid = threadIdx.x >> 5, lane = threadIdx.x & 31;
  int m0 = blockIdx.y * 128 + wid * 16;
  const float* ab = attn + (long)bh * TT * TT;
  const bf16_t* vb = vT + (long)bh * HD * TT;
  floatx8 acc[4] = {};
  int ar = m0 + (lane & 15);
  int kh = (lane >> 4) * 16;
  for (int k0 = 0; k0 < TT; k0 += 32) {
    const float* ap = ab + (long)ar * TT + k0 + kh;
    __builtin_prefetch(ap + 256, 0, 1);   // stream the fp32 attn rows
    bf16x16 a;
#pragma unroll
    for (int i = 0; i < 16; ++i) a[i] = (bf16_t)ap[i];
#pragma unroll
    for (int j = 0; j < 4; ++j) {
      bf16x16 bt = *(const bf16x16*)(vb + (long)(j * 16 + (lane & 15)) * TT + k0 + kh);
      acc[j] = wmma_bf16(a, bt, acc[j]);
    }
  }
  int b = bh / NH, hh = bh % NH;
#pragma unroll
  for (int j = 0; j < 4; ++j) {
    int d = j * 16 + (lane & 15);
#pragma unroll
    for (int e = 0; e < 8; ++e) {
      int m = m0 + e + (lane >> 4) * 8;
      ctx[((long)(b * TT + m)) * DM + hh * HD + d] = (bf16_t)acc[j][e];
    }
  }
}

extern "C" void kernel_launch(void* const* d_in, const int* in_sizes, int n_in,
                              void* d_out, int out_size, void* d_ws, size_t ws_size,
                              hipStream_t stream) {
  const float* x     = (const float*)d_in[0];
  const unsigned char* mask = (const unsigned char*)d_in[1];
  const float* ln1_w = (const float*)d_in[2];
  const float* ln1_b = (const float*)d_in[3];
  const float* Wq    = (const float*)d_in[4];
  const float* bq    = (const float*)d_in[5];
  const float* Wk    = (const float*)d_in[6];
  const float* bk    = (const float*)d_in[7];
  const float* Wv    = (const float*)d_in[8];
  const float* bv    = (const float*)d_in[9];
  const float* Wo    = (const float*)d_in[10];
  const float* bo    = (const float*)d_in[11];
  const float* ln2_w = (const float*)d_in[12];
  const float* ln2_b = (const float*)d_in[13];
  const float* W1    = (const float*)d_in[14];
  const float* b1    = (const float*)d_in[15];
  const float* W2    = (const float*)d_in[16];
  const float* b2    = (const float*)d_in[17];

  float* out_x    = (float*)d_out;
  float* out_attn = (float*)d_out + (size_t)BB * TT * DM;

  char* wp = (char*)d_ws;
  auto alloc = [&](size_t bytes) { char* p = wp; wp += (bytes + 255) & ~(size_t)255; return p; };
  bf16_t* h_bf  = (bf16_t*)alloc((size_t)BB * TT * DM * 2);
  bf16_t* wqT   = (bf16_t*)alloc((size_t)DM * DM * 2);
  bf16_t* wkT   = (bf16_t*)alloc((size_t)DM * DM * 2);
  bf16_t* wvT   = (bf16_t*)alloc((size_t)DM * DM * 2);
  bf16_t* woT   = (bf16_t*)alloc((size_t)DM * DM * 2);
  bf16_t* w1T   = (bf16_t*)alloc((size_t)DM * DFF * 2);
  bf16_t* w2T   = (bf16_t*)alloc((size_t)DFF * DM * 2);
  bf16_t* q_bf  = (bf16_t*)alloc((size_t)BB * TT * DM * 2);
  bf16_t* k_bf  = (bf16_t*)alloc((size_t)BB * TT * DM * 2);
  bf16_t* vT_bf = (bf16_t*)alloc((size_t)BB * TT * DM * 2);
  bf16_t* ctx_bf= (bf16_t*)alloc((size_t)BB * TT * DM * 2);
  float*  x1    = (float*) alloc((size_t)BB * TT * DM * 4);
  bf16_t* h2_bf = (bf16_t*)alloc((size_t)BB * TT * DM * 2);
  bf16_t* ff_bf = (bf16_t*)alloc((size_t)BB * TT * DFF * 2);

  const int M = BB * TT;               // 4096 rows
  const dim3 blk(256);

  // 1) weight transpose+convert
  transpose_bf16_kernel<<<dim3((DM * DM) / 256), blk, 0, stream>>>(Wq, wqT, DM, DM);
  transpose_bf16_kernel<<<dim3((DM * DM) / 256), blk, 0, stream>>>(Wk, wkT, DM, DM);
  transpose_bf16_kernel<<<dim3((DM * DM) / 256), blk, 0, stream>>>(Wv, wvT, DM, DM);
  transpose_bf16_kernel<<<dim3((DM * DM) / 256), blk, 0, stream>>>(Wo, woT, DM, DM);
  transpose_bf16_kernel<<<dim3((DM * DFF) / 256), blk, 0, stream>>>(W1, w1T, DM, DFF);
  transpose_bf16_kernel<<<dim3((DFF * DM) / 256), blk, 0, stream>>>(W2, w2T, DFF, DM);

  // 2) LN1
  ln_bf16_kernel<<<dim3(M / 8), blk, 0, stream>>>(x, ln1_w, ln1_b, h_bf, M);

  // 3) QKV projections (TDM + WMMA), scatter to per-head layouts
  gemm_bf16_kernel<3><<<dim3(DM / 256, M / 32), blk, 0, stream>>>(h_bf, wqT, bq, nullptr, nullptr, q_bf,  M, DM, DM);
  gemm_bf16_kernel<3><<<dim3(DM / 256, M / 32), blk, 0, stream>>>(h_bf, wkT, bk, nullptr, nullptr, k_bf,  M, DM, DM);
  gemm_bf16_kernel<4><<<dim3(DM / 256, M / 32), blk, 0, stream>>>(h_bf, wvT, bv, nullptr, nullptr, vT_bf, M, DM, DM);

  // 4) scores (masked, scaled) -> d_out attn region; softmax in place
  scores_kernel<<<dim3(TT / 256, TT / 32, BB * NH), blk, 0, stream>>>(q_bf, k_bf, mask, out_attn);
  softmax_kernel<<<dim3((BB * NH * TT) / 8), blk, 0, stream>>>(out_attn, BB * NH * TT);

  // 5) ctx = attn @ V
  ctx_kernel<<<dim3(1, TT / 128, BB * NH), blk, 0, stream>>>(out_attn, vT_bf, ctx_bf);

  // 6) output projection + residual -> x1
  gemm_bf16_kernel<2><<<dim3(DM / 256, M / 32), blk, 0, stream>>>(ctx_bf, woT, bo, x, x1, nullptr, M, DM, DM);

  // 7) LN2, FFN (GELU), final projection + residual -> d_out x region
  ln_bf16_kernel<<<dim3(M / 8), blk, 0, stream>>>(x1, ln2_w, ln2_b, h2_bf, M);
  gemm_bf16_kernel<1><<<dim3(DFF / 256, M / 32), blk, 0, stream>>>(h2_bf, w1T, b1, nullptr, nullptr, ff_bf, M, DFF, DM);
  gemm_bf16_kernel<2><<<dim3(DM / 256, M / 32), blk, 0, stream>>>(ff_bf, w2T, b2, x1, out_x, nullptr, M, DM, DFF);
}
